// EmotionAwareTTSModel_20332375179570
// MI455X (gfx1250) — matrix operand order repaired and checked by reading, not compiled, not run on hardware
//
#include <hip/hip_runtime.h>

// ---------------------------------------------------------------------------
// EmotionAwareTTS forward for MI455X (gfx1250, wave32, WMMA 16x16x32 f16)
// All GEMM A/B operands are f16 in global memory -> barrier-free streaming
// WMMA GEMM (no LDS staging).  Producers emit f16 directly.
// ---------------------------------------------------------------------------

typedef __attribute__((ext_vector_type(16))) _Float16 v16h;
typedef __attribute__((ext_vector_type(8)))  _Float16 v8h;
typedef __attribute__((ext_vector_type(8)))  float    v8f;

#define TB   16      // batch
#define TS   128     // text length
#define TE   512     // embed dim
#define THD  512     // encoder hidden (per dir)
#define TH   1024    // decoder hidden
#define DATT 1536
#define NHE  8       // heads
#define DH   192     // head dim
#define ML   1000    // MAXLEN
#define MEL  80

__device__ __forceinline__ float sigmf(float x) { return 1.f / (1.f + __expf(-x)); }

__device__ __forceinline__ v16h make_frag(v8h lo, v8h hi) {
  v16h r;
#pragma unroll
  for (int i = 0; i < 8; i++) { r[i] = lo[i]; r[i + 8] = hi[i]; }
  return r;
}

// ---------------------------------------------------------------------------
// Batched streaming GEMM:  C[M,N] = A[M,K](f16) * Bt[N,K](f16)^T (+bias)(relu)
// Outputs: optional f32 C and/or f16 Ch.  Block = 256 threads = 8 waves;
// block tile 32(M) x 128(N); two 16x16 WMMA tiles per wave sharing one A
// fragment.  No LDS, no barriers: the K loop is pure global_load_b128 + WMMA,
// fully pipelineable.  M must be a multiple of 32 (true for every launch).
// ---------------------------------------------------------------------------
__global__ void k_gemm(const _Float16* __restrict__ A, long long lda, long long sAo, long long sAi,
                       const _Float16* __restrict__ Bt, long long ldb, long long sBo, long long sBi,
                       float* __restrict__ C, _Float16* __restrict__ Ch,
                       long long ldc, long long sCo, long long sCi,
                       const float* __restrict__ bias,
                       int M, int N, int K, int innerN, int relu) {
  const int zo = blockIdx.z / innerN;
  const int zi = blockIdx.z - zo * innerN;
  A  += zo * sAo + zi * sAi;
  Bt += zo * sBo + zi * sBi;
  const long long coff = zo * sCo + zi * sCi;

  const int mBase = blockIdx.y * 32;
  const int nBase = blockIdx.x * 128;
  const int lane  = threadIdx.x & 31;
  const int wave  = threadIdx.x >> 5;
  const int mt    = wave >> 2;                    // 0..1
  const int nt    = wave & 3;                     // 0..3
  const int hi    = lane >> 4;
  const int l15   = lane & 15;

  const int n1  = nBase + nt * 16 + l15;
  const int n2  = n1 + 64;
  const int nc1 = (n1 < N) ? n1 : (N - 1);        // clamp loads, guard stores
  const int nc2 = (n2 < N) ? n2 : (N - 1);
  const _Float16* bp1 = Bt + (long long)nc1 * ldb + hi * 8;
  const _Float16* bp2 = Bt + (long long)nc2 * ldb + hi * 8;
  const _Float16* ap  = A + (long long)(mBase + mt * 16 + l15) * lda + hi * 8;

  v8f acc1 = {}, acc2 = {};
  for (int k0 = 0; k0 < K; k0 += 32) {
    v16h a  = make_frag(*(const v8h*)(ap + k0), *(const v8h*)(ap + k0 + 16));
    v16h b1 = make_frag(*(const v8h*)(bp1 + k0), *(const v8h*)(bp1 + k0 + 16));
    v16h b2 = make_frag(*(const v8h*)(bp2 + k0), *(const v8h*)(bp2 + k0 + 16));
    if (k0 + 32 < K) {
      __builtin_prefetch(bp1 + k0 + 32, 0, 3);    // global_prefetch_b8
      __builtin_prefetch(bp2 + k0 + 32, 0, 3);
    }
    acc1 = __builtin_amdgcn_wmma_f32_16x16x32_f16(false, a, false, b1, (short)0, acc1,
                                                  false, false);
    acc2 = __builtin_amdgcn_wmma_f32_16x16x32_f16(false, a, false, b2, (short)0, acc2,
                                                  false, false);
  }

#pragma unroll
  for (int r = 0; r < 8; r++) {
    const long long mg = mBase + mt * 16 + hi * 8 + r;
    float v1 = acc1[r], v2 = acc2[r];
    if (bias) {
      if (n1 < N) v1 += bias[n1];
      if (n2 < N) v2 += bias[n2];
    }
    if (relu) { v1 = fmaxf(v1, 0.f); v2 = fmaxf(v2, 0.f); }
    const long long row = coff + mg * ldc;
    if (C) {
      if (n1 < N) C[row + n1] = v1;
      if (n2 < N) C[row + n2] = v2;
    }
    if (Ch) {
      if (n1 < N) Ch[row + n1] = (_Float16)v1;
      if (n2 < N) Ch[row + n2] = (_Float16)v2;
    }
  }
}

// ---------------------------------------------------------------------------
// Persistent single-workgroup LSTM recurrence (M = batch = 16 = one WMMA tile).
// 1024 threads = 32 waves.  h (f16) + gates (f16) live in LDS (dec: 160KB,
// needs CDNA5's 320KB WGP LDS).  Cell state c stays in registers.  Per K-chunk
// the A fragment is loaded once and reused across all NT n-tiles.  Hidden
// stream is written to global as f16 (consumers are GEMM-A operands).
// ---------------------------------------------------------------------------
template <int NHID>
__global__ void k_lstm(const float* __restrict__ xw,      // [16*S, 4*NHID], row = b*S + t
                       const _Float16* __restrict__ whT,  // [4*NHID, NHID]
                       _Float16* __restrict__ hout, int ldh, int colOff,
                       int S, int reverse) {
  constexpr int NG = 4 * NHID;
  constexpr int HP = NHID + 8;
  constexpr int NT = NG / (16 * 32);            // n-tiles per wave (4 enc, 8 dec)
  constexpr int NE = (16 * NHID) / 1024;        // elementwise items per thread
  constexpr int SH = (NHID == 512) ? 9 : 10;

  extern __shared__ _Float16 smem[];
  _Float16* hbuf = smem;                        // [16][HP]
  _Float16* gbuf = smem + 16 * HP;              // [16][NG]

  const int lane = threadIdx.x & 31;
  const int wave = threadIdx.x >> 5;
  const int hi   = lane >> 4;
  const int l15  = lane & 15;

  for (int i = threadIdx.x; i < 16 * HP; i += 1024) hbuf[i] = (_Float16)0.f;
  float creg[NE];
#pragma unroll
  for (int e = 0; e < NE; e++) creg[e] = 0.f;
  __syncthreads();

  const _Float16* apb = hbuf + l15 * HP + hi * 8;
  const _Float16* bpw = whT + (long long)(wave * NT * 16 + l15) * NHID + hi * 8;

  for (int t = 0; t < S; t++) {
    const int tt = reverse ? (S - 1 - t) : t;
    // ---- gates = xw[:,tt,:] + h @ Wh^T  (WMMA, A-frag shared across tiles) ----
    v8f accs[NT];
#pragma unroll
    for (int i = 0; i < NT; i++) {
      const int n = (wave * NT + i) * 16 + l15;
#pragma unroll
      for (int r = 0; r < 8; r++) {
        int m = hi * 8 + r;
        accs[i][r] = xw[((long long)(m * S + tt)) * NG + n];
      }
    }
    for (int k0 = 0; k0 < NHID; k0 += 32) {
      v16h a = make_frag(*(const v8h*)(apb + k0), *(const v8h*)(apb + k0 + 16));
#pragma unroll
      for (int i = 0; i < NT; i++) {
        const _Float16* bp = bpw + (long long)i * 16 * NHID + k0;
        v16h b = make_frag(*(const v8h*)bp, *(const v8h*)(bp + 16));
        accs[i] = __builtin_amdgcn_wmma_f32_16x16x32_f16(false, a, false, b, (short)0,
                                                         accs[i], false, false);
      }
    }
#pragma unroll
    for (int i = 0; i < NT; i++) {
      const int n = (wave * NT + i) * 16 + l15;
#pragma unroll
      for (int r = 0; r < 8; r++) {
        int m = hi * 8 + r;
        gbuf[m * NG + n] = (_Float16)accs[i][r];
      }
    }
    __syncthreads();
    // ---- elementwise cell update; c in registers ----
#pragma unroll
    for (int e = 0; e < NE; e++) {
      int idx = threadIdx.x + e * 1024;
      int m   = idx >> SH;
      int col = idx & (NHID - 1);
      float ig = (float)gbuf[m * NG + col];
      float fg = (float)gbuf[m * NG + NHID + col];
      float gg = (float)gbuf[m * NG + 2 * NHID + col];
      float og = (float)gbuf[m * NG + 3 * NHID + col];
      float cs = sigmf(fg) * creg[e] + sigmf(ig) * tanhf(gg);
      float hn = sigmf(og) * tanhf(cs);
      creg[e] = cs;
      _Float16 hh16 = (_Float16)hn;
      hbuf[m * HP + col] = hh16;
      hout[((long long)(m * S + tt)) * (long long)ldh + colOff + col] = hh16;
    }
    __syncthreads();
  }
}

// ---------------------------------------------------------------------------
// Small helper kernels
// ---------------------------------------------------------------------------
__global__ void k_embed(const int* __restrict__ ids, const float* __restrict__ emb,
                        _Float16* __restrict__ te) {
  long long i = (long long)blockIdx.x * blockDim.x + threadIdx.x;
  if (i >= (long long)TB * TS * TE) return;
  int row = (int)(i >> 9);
  int d   = (int)(i & (TE - 1));
  te[i] = (_Float16)emb[(long long)ids[row] * TE + d];
}

__global__ void k_cast(const float* __restrict__ in, _Float16* __restrict__ out, long long n) {
  long long i = (long long)blockIdx.x * blockDim.x + threadIdx.x;
  if (i < n) out[i] = (_Float16)in[i];
}

// f32 [K,N] -> f16 [N,K]
__global__ void k_tcast(const float* __restrict__ in, _Float16* __restrict__ out, int K, int N) {
  long long i = (long long)blockIdx.x * blockDim.x + threadIdx.x;
  if (i >= (long long)K * N) return;
  int nn = (int)(i % N);
  long long kk = i / N;
  out[(long long)nn * K + kk] = (_Float16)in[kk * N + nn];
}

__global__ void k_add2(const float* a, const float* b, float* o, int n) {
  int i = blockIdx.x * blockDim.x + threadIdx.x;
  if (i < n) o[i] = a[i] + b[i];
}

// Emotion MLP (K=5 then K=256) + relu, broadcast into combined[:, :, 1024:1536]
__global__ void k_emo(const float* __restrict__ eemb, const float* __restrict__ w1,
                      const float* __restrict__ b1, const float* __restrict__ w2,
                      const float* __restrict__ b2, _Float16* __restrict__ combined) {
  __shared__ float h1s[16 * 256];
  for (int idx = threadIdx.x; idx < 16 * 256; idx += blockDim.x) {
    int b = idx >> 8, j = idx & 255;
    float acc = b1[j];
#pragma unroll
    for (int k = 0; k < 5; k++) acc += eemb[b * 5 + k] * w1[k * 256 + j];
    h1s[idx] = fmaxf(acc, 0.f);
  }
  __syncthreads();
  for (int idx = threadIdx.x; idx < 16 * 512; idx += blockDim.x) {
    int b = idx >> 9, j = idx & 511;
    float acc = b2[j];
    for (int k = 0; k < 256; k++) acc += h1s[b * 256 + k] * w2[k * 512 + j];
    _Float16 v = (_Float16)fmaxf(acc, 0.f);
    for (int s = 0; s < TS; s++)
      combined[((long long)(b * TS + s)) * DATT + 2 * THD + j] = v;
  }
}

// split K/V (f16 [B*S, DATT]) into per-head buffers: kh[bh][s][d], vT[bh][d][s]
__global__ void k_split_heads(const _Float16* __restrict__ kb, const _Float16* __restrict__ vb,
                              _Float16* __restrict__ kh, _Float16* __restrict__ vT) {
  long long i = (long long)blockIdx.x * blockDim.x + threadIdx.x;
  if (i >= (long long)TB * NHE * TS * DH) return;
  int d = (int)(i % DH);
  long long r = i / DH;
  int s = (int)(r % TS); r /= TS;
  int h = (int)(r % NHE);
  int b = (int)(r / NHE);
  long long src = ((long long)(b * TS + s)) * DATT + h * DH + d;
  kh[(((long long)(b * NHE + h) * TS) + s) * DH + d] = kb[src];
  vT[(((long long)(b * NHE + h) * DH) + d) * TS + s] = vb[src];
}

// softmax over rows of 128, wave-per-row, scale applied first; f16 copy out
__global__ void k_softmax(float* __restrict__ p, _Float16* __restrict__ ph,
                          int rows, float scale) {
  int wave = threadIdx.x >> 5, lane = threadIdx.x & 31;
  int row = blockIdx.x * (blockDim.x >> 5) + wave;
  if (row >= rows) return;
  float* r = p + (long long)row * 128;
  _Float16* rh = ph + (long long)row * 128;
  float v[4];
  float mx = -1e30f;
#pragma unroll
  for (int i = 0; i < 4; i++) { v[i] = r[lane + i * 32] * scale; mx = fmaxf(mx, v[i]); }
  for (int o = 16; o > 0; o >>= 1) mx = fmaxf(mx, __shfl_xor(mx, o, 32));
  float sum = 0.f;
#pragma unroll
  for (int i = 0; i < 4; i++) { v[i] = __expf(v[i] - mx); sum += v[i]; }
  for (int o = 16; o > 0; o >>= 1) sum += __shfl_xor(sum, o, 32);
  float inv = 1.f / sum;
#pragma unroll
  for (int i = 0; i < 4; i++) {
    float q = v[i] * inv;
    r[lane + i * 32]  = q;
    rh[lane + i * 32] = (_Float16)q;
  }
}

// variance head layer-2 (N=1): wave-per-row dot product
__global__ void k_head2(const float* __restrict__ hh, const float* __restrict__ w2,
                        const float* __restrict__ b2, float* __restrict__ out,
                        int rows, int Kd, int relu) {
  int wave = threadIdx.x >> 5, lane = threadIdx.x & 31;
  int row = blockIdx.x * (blockDim.x >> 5) + wave;
  if (row >= rows) return;
  const float* hr = hh + (long long)row * Kd;
  float acc = 0.f;
  for (int k = lane; k < Kd; k += 32) acc += hr[k] * w2[k];
  for (int o = 16; o > 0; o >>= 1) acc += __shfl_xor(acc, o, 32);
  if (lane == 0) {
    float v = acc + b2[0];
    if (relu) v = fmaxf(v, 0.f);
    out[row] = v;
  }
}

// length regulator: one block per batch; reads f32 attended, writes f16 expanded
__global__ void k_lenreg(const float* __restrict__ att, const float* __restrict__ dur,
                         _Float16* __restrict__ expanded) {
  __shared__ int csum[TS];
  __shared__ int idxs[ML];
  const int b = blockIdx.x;
  if (threadIdx.x == 0) {
    int acc = 0;
    for (int i = 0; i < TS; i++) {
      int r = (int)dur[b * TS + i];              // trunc toward zero, dur >= 0
      r = (r < 1) ? 1 : ((r > ML / TS) ? ML / TS : r);
      acc += r;
      csum[i] = acc;
    }
  }
  __syncthreads();
  const int total = csum[TS - 1];
  for (int t = threadIdx.x; t < ML; t += blockDim.x) {
    int lo = 0, hi2 = TS;
    while (lo < hi2) {                           // searchsorted 'right'
      int mid = (lo + hi2) >> 1;
      if (csum[mid] <= t) lo = mid + 1; else hi2 = mid;
    }
    int idx = (lo > TS - 1) ? TS - 1 : lo;
    idxs[t] = (t < total) ? idx : -1;
  }
  __syncthreads();
  for (long long e = threadIdx.x; e < (long long)ML * DATT; e += blockDim.x) {
    int t = (int)(e / DATT);
    int d = (int)(e % DATT);
    int idx = idxs[t];
    float v = (idx >= 0) ? att[((long long)(b * TS + idx)) * DATT + d] : 0.f;
    expanded[((long long)(b * ML) + t) * DATT + d] = (_Float16)v;
  }
}

// ---------------------------------------------------------------------------
// Host orchestration
// ---------------------------------------------------------------------------
static inline void gemm(hipStream_t s,
                        const _Float16* A, long long lda, long long sAo, long long sAi,
                        const _Float16* Bt, long long ldb, long long sBo, long long sBi,
                        float* C, _Float16* Ch, long long ldc, long long sCo, long long sCi,
                        const float* bias, int M, int N, int K,
                        int batches, int innerN, int relu) {
  dim3 g((N + 127) / 128, (M + 31) / 32, batches);
  k_gemm<<<g, 256, 0, s>>>(A, lda, sAo, sAi, Bt, ldb, sBo, sBi, C, Ch, ldc, sCo, sCi,
                           bias, M, N, K, innerN, relu);
}

extern "C" void kernel_launch(void* const* d_in, const int* in_sizes, int n_in,
                              void* d_out, int out_size, void* d_ws, size_t ws_size,
                              hipStream_t stream) {
  (void)in_sizes; (void)n_in; (void)out_size; (void)ws_size;
  // -------- inputs (setup_inputs order; params dict insertion order) --------
  const int*   ids   = (const int*)d_in[0];
  const float* eemb  = (const float*)d_in[1];
  const float* emb   = (const float*)d_in[2];
  const float* efwi  = (const float*)d_in[3];
  const float* efwh  = (const float*)d_in[4];
  const float* efbi  = (const float*)d_in[5];
  const float* efbh  = (const float*)d_in[6];
  const float* ebwi  = (const float*)d_in[7];
  const float* ebwh  = (const float*)d_in[8];
  const float* ebbi  = (const float*)d_in[9];
  const float* ebbh  = (const float*)d_in[10];
  const float* ew1   = (const float*)d_in[11];
  const float* eb1   = (const float*)d_in[12];
  const float* ew2   = (const float*)d_in[13];
  const float* eb2   = (const float*)d_in[14];
  const float* wq    = (const float*)d_in[15];
  const float* bq    = (const float*)d_in[16];
  const float* wk    = (const float*)d_in[17];
  const float* bk    = (const float*)d_in[18];
  const float* wv    = (const float*)d_in[19];
  const float* bv    = (const float*)d_in[20];
  const float* wo    = (const float*)d_in[21];
  const float* bo    = (const float*)d_in[22];
  const float* durw1 = (const float*)d_in[23];
  const float* durb1 = (const float*)d_in[24];
  const float* durw2 = (const float*)d_in[25];
  const float* durb2 = (const float*)d_in[26];
  const float* pitw1 = (const float*)d_in[27];
  const float* pitb1 = (const float*)d_in[28];
  const float* pitw2 = (const float*)d_in[29];
  const float* pitb2 = (const float*)d_in[30];
  const float* enew1 = (const float*)d_in[31];
  const float* eneb1 = (const float*)d_in[32];
  const float* enew2 = (const float*)d_in[33];
  const float* eneb2 = (const float*)d_in[34];
  const float* d1wi  = (const float*)d_in[35];
  const float* d1wh  = (const float*)d_in[36];
  const float* d1bi  = (const float*)d_in[37];
  const float* d1bh  = (const float*)d_in[38];
  const float* d2wi  = (const float*)d_in[39];
  const float* d2wh  = (const float*)d_in[40];
  const float* d2bi  = (const float*)d_in[41];
  const float* d2bh  = (const float*)d_in[42];
  const float* melw  = (const float*)d_in[43];
  const float* melb  = (const float*)d_in[44];

  // -------- d_out layout: mel | dur | pitch | energy | attended --------
  float* out  = (float*)d_out;
  float* melO = out;                                   // 16*1000*80
  float* durO = out + (long long)TB * ML * MEL;        // 2048
  float* pitO = durO + TB * TS;
  float* eneO = pitO + TB * TS;
  float* attO = eneO + TB * TS;                        // 16*128*1536

  // -------- workspace carve-out --------
  char* wp = (char*)d_ws;
  auto alloc = [&](size_t bytes) -> void* {
    void* r = (void*)wp;
    wp += (bytes + 255) & ~(size_t)255;
    return r;
  };
  _Float16* teh   = (_Float16*)alloc((size_t)TB * TS * TE * 2);
  _Float16* wiFf  = (_Float16*)alloc((size_t)4 * THD * TE * 2);
  _Float16* whFf  = (_Float16*)alloc((size_t)4 * THD * THD * 2);
  _Float16* wiBf  = (_Float16*)alloc((size_t)4 * THD * TE * 2);
  _Float16* whBf  = (_Float16*)alloc((size_t)4 * THD * THD * 2);
  float*    bsF   = (float*)alloc((size_t)4 * THD * 4);
  float*    bsB   = (float*)alloc((size_t)4 * THD * 4);
  float*    xwF   = (float*)alloc((size_t)TB * TS * 4 * THD * 4);
  float*    xwB   = (float*)alloc((size_t)TB * TS * 4 * THD * 4);
  _Float16* combh = (_Float16*)alloc((size_t)TB * TS * DATT * 2);
  _Float16* wqT   = (_Float16*)alloc((size_t)DATT * DATT * 2);
  _Float16* wkT   = (_Float16*)alloc((size_t)DATT * DATT * 2);
  _Float16* wvT   = (_Float16*)alloc((size_t)DATT * DATT * 2);
  _Float16* woT   = (_Float16*)alloc((size_t)DATT * DATT * 2);
  _Float16* qh    = (_Float16*)alloc((size_t)TB * TS * DATT * 2);
  _Float16* khf   = (_Float16*)alloc((size_t)TB * TS * DATT * 2);
  _Float16* vhf   = (_Float16*)alloc((size_t)TB * TS * DATT * 2);
  _Float16* khh   = (_Float16*)alloc((size_t)TB * NHE * TS * DH * 2);
  _Float16* vTh   = (_Float16*)alloc((size_t)TB * NHE * DH * TS * 2);
  float*    probs = (float*)alloc((size_t)TB * NHE * TS * TS * 4);
  _Float16* probh = (_Float16*)alloc((size_t)TB * NHE * TS * TS * 2);
  _Float16* ctxh  = (_Float16*)alloc((size_t)TB * TS * DATT * 2);
  _Float16* atth  = (_Float16*)alloc((size_t)TB * TS * DATT * 2);
  _Float16* dw1T  = (_Float16*)alloc((size_t)THD * DATT * 2);
  _Float16* pw1T  = (_Float16*)alloc((size_t)THD * DATT * 2);
  _Float16* xw1T  = (_Float16*)alloc((size_t)THD * DATT * 2);   // energy w1T
  float*    hh    = (float*)alloc((size_t)TB * TS * THD * 4);
  _Float16* expdh = (_Float16*)alloc((size_t)TB * ML * DATT * 2);
  _Float16* wi1f  = (_Float16*)alloc((size_t)4 * TH * DATT * 2);
  _Float16* wh1f  = (_Float16*)alloc((size_t)4 * TH * TH * 2);
  _Float16* wi2f  = (_Float16*)alloc((size_t)4 * TH * TH * 2);
  _Float16* wh2f  = (_Float16*)alloc((size_t)4 * TH * TH * 2);
  float*    bs1   = (float*)alloc((size_t)4 * TH * 4);
  float*    bs2   = (float*)alloc((size_t)4 * TH * 4);
  float*    xwD   = (float*)alloc((size_t)TB * ML * 4 * TH * 4); // reused dec1/dec2
  _Float16* h1h   = (_Float16*)alloc((size_t)TB * ML * TH * 2);
  _Float16* h2h   = (_Float16*)alloc((size_t)TB * ML * TH * 2);
  _Float16* melT  = (_Float16*)alloc((size_t)MEL * TH * 2);

  const hipStream_t s = stream;
  auto castN = [&](const float* in, _Float16* o, long long n) {
    k_cast<<<(unsigned)((n + 255) / 256), 256, 0, s>>>(in, o, n);
  };
  auto tcastN = [&](const float* in, _Float16* o, int K, int N) {
    long long n = (long long)K * N;
    k_tcast<<<(unsigned)((n + 255) / 256), 256, 0, s>>>(in, o, K, N);
  };

  // ============ encoder ============
  k_embed<<<(TB * TS * TE + 255) / 256, 256, 0, s>>>(ids, emb, teh);
  castN(efwi, wiFf, (long long)4 * THD * TE);
  castN(efwh, whFf, (long long)4 * THD * THD);
  castN(ebwi, wiBf, (long long)4 * THD * TE);
  castN(ebwh, whBf, (long long)4 * THD * THD);
  k_add2<<<(4 * THD + 255) / 256, 256, 0, s>>>(efbi, efbh, bsF, 4 * THD);
  k_add2<<<(4 * THD + 255) / 256, 256, 0, s>>>(ebbi, ebbh, bsB, 4 * THD);
  gemm(s, teh, TE, 0, 0, wiFf, TE, 0, 0, xwF, nullptr, 4 * THD, 0, 0, bsF,
       TB * TS, 4 * THD, TE, 1, 1, 0);
  gemm(s, teh, TE, 0, 0, wiBf, TE, 0, 0, xwB, nullptr, 4 * THD, 0, 0, bsB,
       TB * TS, 4 * THD, TE, 1, 1, 0);
  const size_t lds_enc = (size_t)(16 * (THD + 8) + 16 * 4 * THD) * sizeof(_Float16);
  k_lstm<THD><<<1, 1024, lds_enc, s>>>(xwF, whFf, combh, DATT, 0, TS, 0);
  k_lstm<THD><<<1, 1024, lds_enc, s>>>(xwB, whBf, combh, DATT, THD, TS, 1);
  k_emo<<<1, 512, 0, s>>>(eemb, ew1, eb1, ew2, eb2, combh);

  // ============ attention ============
  tcastN(wq, wqT, DATT, DATT);
  tcastN(wk, wkT, DATT, DATT);
  tcastN(wv, wvT, DATT, DATT);
  tcastN(wo, woT, DATT, DATT);
  gemm(s, combh, DATT, 0, 0, wqT, DATT, 0, 0, nullptr, qh, DATT, 0, 0, bq,
       TB * TS, DATT, DATT, 1, 1, 0);
  gemm(s, combh, DATT, 0, 0, wkT, DATT, 0, 0, nullptr, khf, DATT, 0, 0, bk,
       TB * TS, DATT, DATT, 1, 1, 0);
  gemm(s, combh, DATT, 0, 0, wvT, DATT, 0, 0, nullptr, vhf, DATT, 0, 0, bv,
       TB * TS, DATT, DATT, 1, 1, 0);
  {
    long long n = (long long)TB * NHE * TS * DH;
    k_split_heads<<<(unsigned)((n + 255) / 256), 256, 0, s>>>(khf, vhf, khh, vTh);
  }
  // scores[bh] = q_head @ k_head^T  (batched over B*NH = 128)
  gemm(s, qh, DATT, (long long)TS * DATT, DH,
       khh, DH, (long long)NHE * TS * DH, (long long)TS * DH,
       probs, nullptr, TS, (long long)NHE * TS * TS, (long long)TS * TS,
       nullptr, TS, TS, DH, TB * NHE, NHE, 0);
  k_softmax<<<(TB * NHE * TS + 7) / 8, 256, 0, s>>>(probs, probh, TB * NHE * TS,
                                                    0.07216878364870323f);  // 1/sqrt(192)
  // ctx[bh] = probs @ v_head   (Bt = vT[bh] : [DH, TS])
  gemm(s, probh, TS, (long long)NHE * TS * TS, (long long)TS * TS,
       vTh, TS, (long long)NHE * DH * TS, (long long)DH * TS,
       nullptr, ctxh, DATT, (long long)TS * DATT, DH,
       nullptr, TS, DH, TS, TB * NHE, NHE, 0);
  gemm(s, ctxh, DATT, 0, 0, woT, DATT, 0, 0, attO, atth, DATT, 0, 0, bo,
       TB * TS, DATT, DATT, 1, 1, 0);

  // ============ variance heads ============
  tcastN(durw1, dw1T, DATT, THD);
  tcastN(pitw1, pw1T, DATT, THD);
  tcastN(enew1, xw1T, DATT, THD);
  gemm(s, atth, DATT, 0, 0, dw1T, DATT, 0, 0, hh, nullptr, THD, 0, 0, durb1,
       TB * TS, THD, DATT, 1, 1, 1);
  k_head2<<<(TB * TS + 7) / 8, 256, 0, s>>>(hh, durw2, durb2, durO, TB * TS, THD, 1);
  gemm(s, atth, DATT, 0, 0, pw1T, DATT, 0, 0, hh, nullptr, THD, 0, 0, pitb1,
       TB * TS, THD, DATT, 1, 1, 1);
  k_head2<<<(TB * TS + 7) / 8, 256, 0, s>>>(hh, pitw2, pitb2, pitO, TB * TS, THD, 0);
  gemm(s, atth, DATT, 0, 0, xw1T, DATT, 0, 0, hh, nullptr, THD, 0, 0, eneb1,
       TB * TS, THD, DATT, 1, 1, 1);
  k_head2<<<(TB * TS + 7) / 8, 256, 0, s>>>(hh, enew2, eneb2, eneO, TB * TS, THD, 0);

  // ============ length regulation ============
  k_lenreg<<<TB, 256, 0, s>>>(attO, durO, expdh);

  // ============ decoder ============
  castN(d1wi, wi1f, (long long)4 * TH * DATT);
  castN(d1wh, wh1f, (long long)4 * TH * TH);
  castN(d2wi, wi2f, (long long)4 * TH * TH);
  castN(d2wh, wh2f, (long long)4 * TH * TH);
  k_add2<<<(4 * TH + 255) / 256, 256, 0, s>>>(d1bi, d1bh, bs1, 4 * TH);
  k_add2<<<(4 * TH + 255) / 256, 256, 0, s>>>(d2bi, d2bh, bs2, 4 * TH);

  const size_t lds_dec = (size_t)(16 * (TH + 8) + 16 * 4 * TH) * sizeof(_Float16);
  gemm(s, expdh, DATT, 0, 0, wi1f, DATT, 0, 0, xwD, nullptr, 4 * TH, 0, 0, bs1,
       TB * ML, 4 * TH, DATT, 1, 1, 0);
  k_lstm<TH><<<1, 1024, lds_dec, s>>>(xwD, wh1f, h1h, TH, 0, ML, 0);
  gemm(s, h1h, TH, 0, 0, wi2f, TH, 0, 0, xwD, nullptr, 4 * TH, 0, 0, bs2,
       TB * ML, 4 * TH, TH, 1, 1, 0);
  k_lstm<TH><<<1, 1024, lds_dec, s>>>(xwD, wh2f, h2h, TH, 0, ML, 0);

  // ============ mel projection ============
  tcastN(melw, melT, TH, MEL);
  gemm(s, h2h, TH, 0, 0, melT, TH, 0, 0, melO, nullptr, MEL, 0, 0, melb,
       TB * ML, MEL, TH, 1, 1, 0);
}
